// Atoms_25125558682029
// MI455X (gfx1250) — matrix operand order repaired and checked by reading, not compiled, not run on hardware
//
#include <hip/hip_runtime.h>

typedef __attribute__((ext_vector_type(16))) _Float16 v16h;
typedef __attribute__((ext_vector_type(8)))  _Float16 v8h;
typedef __attribute__((ext_vector_type(8)))  float    v8f;

#define PI_F 3.14159265358979323846f

#define N_SAMPLES 32768
#define N_EVENTS  16
#define WINDOW    512
#define STEP      256
#define N_FRAMES  128
#define NBINS     257
#define NPAD      272          // 257 padded to 17*16
#define NB2       544          // [Re(0..256) pad | Im(0..256) pad]
#define NBATCH    32
#define NROWS     512          // batch*events
#define NFR_TOT   65536        // NROWS*N_FRAMES
#define SIGPAD    33024        // (N_FRAMES+1)*STEP
#define N2        16384        // complex FFT length for real-32768
#define LOGN2     14

#define NT_FWD    34           // 544/16 N tiles, forward DFT
#define KB_FWD    16           // 512/32 K steps, forward DFT
#define NT_INV    32           // 512/16 N tiles, inverse DFT
#define KB_INV    17           // 544/32 K steps, inverse DFT
#define FRAG      512          // halves per (ntile,ktile) fragment: 32 lanes * 16
#define LDSA_STR  520          // padded LDS row stride (halves), fwd kernel
#define LDSI_STR  552          // padded LDS row stride (halves), inv kernel

__device__ __forceinline__ float clip01(float v) { return fminf(fmaxf(v, 0.f), 1.f); }

// Combine two contiguous 8-half LDS chunks into a WMMA A fragment.
__device__ __forceinline__ v16h load_a_frag(const _Float16* p) {
    v8h lo = *(const v8h*)p;
    v8h hi = *(const v8h*)(p + 16);
    return __builtin_shufflevector(lo, hi, 0,1,2,3,4,5,6,7,8,9,10,11,12,13,14,15);
}

// K-pipelined WMMA tile: 3-deep rotating B buffer keeps two fragments
// (4x global_load_b128) in flight past the one being consumed.
template<int KB>
__device__ __forceinline__ v8f wmma_tile(const v16h* aF, const _Float16* bptr) {
    v8f c = {};
    v16h bbuf[3];
    bbuf[0] = *(const v16h*)(bptr);
    bbuf[1] = *(const v16h*)(bptr + FRAG);
#pragma unroll
    for (int kb = 0; kb < KB; ++kb) {
        if (kb + 2 < KB)
            bbuf[(kb + 2) % 3] = *(const v16h*)(bptr + (size_t)(kb + 2) * FRAG);
        c = __builtin_amdgcn_wmma_f32_16x16x32_f16(false, aF[kb], false, bbuf[kb % 3],
                                                   (short)0, c, false, false);
    }
    return c;
}

// ---------------------------------------------------------------------------
// Table builder: DFT matrices pre-swizzled into WMMA B-fragment order:
//   frag[(nt*KB + kb)*512 + lane*16 + h]  holds  B[k, n]
//   with k = kb*32 + (h>=8 ? 16 : 0) + (lane>=16 ? 8 : 0) + (h & 7),  n = nt*16 + (lane & 15)
// ---------------------------------------------------------------------------
__global__ void k_build_tables(_Float16* __restrict__ BfwdF, _Float16* __restrict__ BirfF) {
    int idx = blockIdx.x * blockDim.x + threadIdx.x;
    const float rs = 0.044194173824159216f;      // 1/sqrt(512)
    const float w0 = 2.0f * PI_F / 512.0f;
    const int total1 = NT_FWD * KB_FWD * FRAG;   // 278528
    if (idx < total1) {
        int h    = idx & 15;
        int lane = (idx >> 4) & 31;
        int kb   = (idx >> 9) & (KB_FWD - 1);
        int nt   = idx >> 13;
        int j    = kb * 32 + ((h >> 3) << 4) + ((lane >> 4) << 3) + (h & 7);   // time sample (K dim)
        int n    = nt * 16 + (lane & 15);              // output column
        float v = 0.f;
        if (n < NBINS)                        v =  __cosf(w0 * (float)j * (float)n) * rs;
        else if (n >= NPAD && n < NPAD+NBINS) v = -__sinf(w0 * (float)j * (float)(n - NPAD)) * rs;
        BfwdF[idx] = (_Float16)v;
    } else {
        int idx2 = idx - total1;
        if (idx2 < NT_INV * KB_INV * FRAG) {
            int h    = idx2 & 15;
            int lane = (idx2 >> 4) & 31;
            int kb   = (idx2 >> 9) % KB_INV;
            int nt   = (idx2 >> 9) / KB_INV;
            int k    = kb * 32 + ((h >> 3) << 4) + ((lane >> 4) << 3) + (h & 7);   // packed spec index (K dim)
            int j    = nt * 16 + (lane & 15);              // output sample
            float v = 0.f;
            if (k < NBINS) {
                float w = (k == 0 || k == 256) ? 1.f : 2.f;
                v = w * __cosf(w0 * (float)k * (float)j) * rs;
            } else if (k >= NPAD && k < NPAD+NBINS) {
                int kk = k - NPAD;
                float w = (kk == 0 || kk == 256) ? 1.f : 2.f;
                v = -w * __sinf(w0 * (float)kk * (float)j) * rs;
            }
            BirfF[idx2] = (_Float16)v;
        }
    }
}

__global__ void k_zero(float* __restrict__ p, int n) {
    int i = blockIdx.x * blockDim.x + threadIdx.x;
    if (i < n) p[i] = 0.f;
}

// ---------------------------------------------------------------------------
// In-LDS iterative radix-2 FFT, 16384 complex points, 512 threads.
// ---------------------------------------------------------------------------
__device__ void block_fft(float* re, float* im, float sign, int tid, int nthr) {
    for (int i = tid; i < N2; i += nthr) {
        unsigned j = __brev((unsigned)i) >> (32 - LOGN2);
        if (j > (unsigned)i) {
            float tr = re[i]; re[i] = re[j]; re[j] = tr;
            float ti = im[i]; im[i] = im[j]; im[j] = ti;
        }
    }
    __syncthreads();
    for (int s = 1; s <= LOGN2; ++s) {
        int half = 1 << (s - 1);
        float base = sign * PI_F / (float)half;   // sign * 2*pi/m
        for (int b = tid; b < (N2 >> 1); b += nthr) {
            int grp = b >> (s - 1);
            int pos = b & (half - 1);
            int i0 = (grp << s) + pos;
            int i1 = i0 + half;
            float c, sn;
            __sincosf(base * (float)pos, &sn, &c);
            float r1 = re[i1], q1 = im[i1];
            float tr = c * r1 - sn * q1;
            float ti = c * q1 + sn * r1;
            float r0 = re[i0], q0 = im[i0];
            re[i1] = r0 - tr; im[i1] = q0 - ti;
            re[i0] = r0 + tr; im[i0] = q0 + ti;
        }
        __syncthreads();
    }
}

__device__ __forceinline__ float shape16(const float* c, int k) {
    // linear_interp_last of 16 coeffs to 16385, align_corners=False
    float pos = ((float)k + 0.5f) * (16.0f / 16385.0f) - 0.5f;
    pos = fminf(fmaxf(pos, 0.f), 15.f);
    int i0 = (int)floorf(pos);
    int i1 = i0 + 1; if (i1 > 15) i1 = 15;
    float w = pos - (float)i0;
    return c[i0] * (1.f - w) + c[i1] * w;
}

// ---------------------------------------------------------------------------
// Kernel 2: spectral noise shaping + Gaussian envelope -> atoms (one row/block)
// Dynamic LDS: 2 * 16384 floats = 128 KB (fits CDNA5 320KB/WGP)
// ---------------------------------------------------------------------------
__global__ __launch_bounds__(512)
void k_noise_atoms(const float* __restrict__ x, const float* __restrict__ noise,
                   float* __restrict__ atoms) {
    extern __shared__ float smem[];
    float* re = smem;
    float* im = smem + N2;
    __shared__ float red[17];
    __shared__ float coeff[16];
    const int row = blockIdx.x;
    const int tid = threadIdx.x;
    const int nthr = blockDim.x;
    const float* xr = x + row * 533;
    const float* nz = noise + (size_t)row * N_SAMPLES;

    if (tid < 16) coeff[tid] = clip01(xr[517 + tid]);
    for (int n = tid; n < N2; n += nthr) {
        re[n] = 2.f * nz[2*n]     - 1.f;
        im[n] = 2.f * nz[2*n + 1] - 1.f;
    }
    __syncthreads();
    block_fft(re, im, -1.f, tid, nthr);

    const float invN = 1.f / 32768.f;
    for (int k = tid; k <= N2 / 2; k += nthr) {
        if (k == 0) {
            float zr = re[0], zi = im[0];
            float U0 = (zr + zi) * shape16(coeff, 0)     * invN;
            float UN = (zr - zi) * shape16(coeff, 16384) * invN;
            re[0] = 0.5f * (U0 + UN);
            im[0] = 0.5f * (U0 - UN);
        } else {
            int km = N2 - k;
            float ar = re[k],  ai = im[k];
            float br = re[km], bi = im[km];
            // unpack rfft(32768) bins k and km from Z[k], Z[km]
            float Er  = 0.5f*(ar+br), Ei  = 0.5f*(ai-bi);
            float Or  = 0.5f*(ai+bi), Oi  = -0.5f*(ar-br);
            float tc, ts; __sincosf(-PI_F * (float)k / (float)N2, &ts, &tc);
            float Ur = Er + tc*Or - ts*Oi;
            float Ui = Ei + tc*Oi + ts*Or;
            float Er2 = 0.5f*(br+ar), Ei2 = 0.5f*(bi-ai);
            float Or2 = 0.5f*(bi+ai), Oi2 = -0.5f*(br-ar);
            float tc2, ts2; __sincosf(-PI_F * (float)km / (float)N2, &ts2, &tc2);
            float Vr = Er2 + tc2*Or2 - ts2*Oi2;
            float Vi = Ei2 + tc2*Oi2 + ts2*Or2;
            // multiply by spec_shape (and 1/N for fwd*inv ortho pair)
            float s1 = shape16(coeff, k)  * invN;
            float s2 = shape16(coeff, km) * invN;
            Ur *= s1; Ui *= s1; Vr *= s2; Vi *= s2;
            // repack for inverse complex FFT
            {
                float sr = Ur + Vr, si = Ui - Vi;
                float dr = Ur - Vr, di = Ui + Vi;
                float ec, es; __sincosf(PI_F * (float)k / (float)N2, &es, &ec);
                re[k] = 0.5f*sr + 0.5f*(-es*dr - ec*di);
                im[k] = 0.5f*si + 0.5f*( ec*dr - es*di);
            }
            {
                float sr = Vr + Ur, si = Vi - Ui;
                float dr = Vr - Ur, di = Vi + Ui;
                float ec, es; __sincosf(PI_F * (float)km / (float)N2, &es, &ec);
                re[km] = 0.5f*sr + 0.5f*(-es*dr - ec*di);
                im[km] = 0.5f*si + 0.5f*( ec*dr - es*di);
            }
        }
    }
    __syncthreads();
    block_fft(re, im, 1.f, tid, nthr);

    // Gaussian envelope with per-row max-norm
    float x0 = clip01(xr[0]), x1 = clip01(xr[1]), amp = clip01(xr[2]);
    float mu    = fminf(fmaxf((x0*2.f - 1.f) * 32768.f, -16384.f), 49152.f);
    float sigma = fminf(fmaxf((1e-8f + x1*0.1f) * 32768.f, 0.f), 32767.f);
    float logs  = __logf(sigma);
    const float c0 = 0.9189385332046727f;   // 0.5*log(2*pi)
    float lmax = -1e30f;
    for (int t = tid; t < N_SAMPLES; t += nthr) {
        float d = ((float)t - mu) / sigma;
        lmax = fmaxf(lmax, -0.5f*d*d - logs - c0);
    }
    for (int o = 16; o > 0; o >>= 1) lmax = fmaxf(lmax, __shfl_xor(lmax, o, 32));
    if ((tid & 31) == 0) red[tid >> 5] = lmax;
    __syncthreads();
    if (tid < 16) {
        float v = red[tid];
        for (int o = 8; o > 0; o >>= 1) v = fmaxf(v, __shfl_xor(v, o, 32));
        if (tid == 0) red[16] = v;
    }
    __syncthreads();
    float pmax = __expf(red[16]);
    float invm = 1.f / (pmax + 1e-8f);
    for (int t = tid; t < N_SAMPLES; t += nthr) {
        float d  = ((float)t - mu) / sigma;
        float p  = __expf(-0.5f*d*d - logs - c0) * invm;
        float nb = (t & 1) ? im[t >> 1] : re[t >> 1];
        atoms[(size_t)row * N_SAMPLES + t] = p * nb * amp;
    }
}

// ---------------------------------------------------------------------------
// Kernel 3: windowed STFT as WMMA GEMM. Block = one 16-row M tile, 8 waves
// sweep the 34 N tiles. A fragments preloaded to registers; B fragments
// 3-deep pipelined 32B/lane vector loads so WMMA overlaps two loads.
// ---------------------------------------------------------------------------
__global__ __launch_bounds__(256)
void k_stft(const float* __restrict__ atoms, const _Float16* __restrict__ BfwdF,
            float* __restrict__ specP) {
    __shared__ _Float16 ldsA[16 * LDSA_STR];
    const int r0  = blockIdx.x * 16;
    const int be  = r0 / N_FRAMES;     // tile never crosses an event row (128%16==0)
    const int tid = threadIdx.x;

    for (int idx = tid; idx < 16 * 512; idx += 256) {
        int m = idx >> 9, j = idx & 511;
        int f = (r0 + m) & (N_FRAMES - 1);
        int t = f * STEP + j;
        float v = 0.f;
        if (t < N_SAMPLES) {
            float ham = 0.54f - 0.46f * __cosf(2.f * PI_F * (float)j / 512.f);
            v = atoms[(size_t)be * N_SAMPLES + t] * ham;
        }
        ldsA[m * LDSA_STR + j] = (_Float16)v;
    }
    __syncthreads();

    const int lane = tid & 31, wave = tid >> 5;
    const int Mrow = lane & 15;
    const int hiA  = (lane >= 16) ? 8 : 0;
    const int hiM  = (lane >= 16) ? 8 : 0;
    const _Float16* ldsRow = ldsA + Mrow * LDSA_STR + hiA;

    v16h aF[KB_FWD];
#pragma unroll
    for (int kb = 0; kb < KB_FWD; ++kb) aF[kb] = load_a_frag(ldsRow + kb * 32);

    for (int nt = wave; nt < NT_FWD; nt += 8) {
        const _Float16* bptr = BfwdF + ((size_t)nt * KB_FWD) * FRAG + lane * 16;
        // near-scope speculative prefetch of next tile group (overrun stays in ws)
        __builtin_prefetch(bptr + (size_t)8 * KB_FWD * FRAG, 0, 3);
        v8f c = wmma_tile<KB_FWD>(aF, bptr);
        const int col = nt * 16 + Mrow;
#pragma unroll
        for (int v = 0; v < 8; ++v) {
            size_t r = (size_t)(r0 + v + hiM);
            specP[r * NB2 + col] = c[v];      // pad columns carry garbage, never read
        }
    }
}

// ---------------------------------------------------------------------------
// Kernel 4: sequential 128-frame resonance scan per (row, bin); writes the
// packed f16 A-matrix [Re|Im] (65536 x 544) for the inverse DFT GEMM.
// ---------------------------------------------------------------------------
__global__ void k_scan(const float* __restrict__ x,
                       const float* __restrict__ specP,
                       _Float16* __restrict__ Afin) {
    int idx = blockIdx.x * blockDim.x + threadIdx.x;   // row*NPAD + k
    if (idx >= NROWS * NPAD) return;
    int row = idx / NPAD, k = idx % NPAD;
    size_t base = (size_t)row * N_FRAMES;
    if (k >= NBINS) {                                   // zero pad lanes
        for (int f = 0; f < N_FRAMES; ++f) {
            size_t o = (base + f) * NB2;
            Afin[o + k]        = (_Float16)0.f;
            Afin[o + NPAD + k] = (_Float16)0.f;
        }
        return;
    }
    const float* xr = x + row * 533;
    float rm   = 0.01f + 0.99f * clip01(xr[3 + k]);
    float freq = PI_F * (float)k / 512.f;
    float rp   = clip01(xr[260 + k]) * (2.f * PI_F) - PI_F + freq;
    float carry = 0.f;
    for (int f = 0; f < N_FRAMES; ++f) {
        size_t si = (base + f) * NB2;
        float rev = specP[si + k];
        float imv = specP[si + NPAD + k];
        float mag = sqrtf(rev*rev + imv*imv) + 1e-8f;
        float ph  = imv / mag * PI_F;
        float m   = mag + rm * carry; carry = m;
        if (f > 0) ph += rp;
        float cc, ss; __sincosf(ph, &ss, &cc);
        Afin[si + k]        = (_Float16)(m * cc);
        Afin[si + NPAD + k] = (_Float16)(m * ss);
    }
}

// ---------------------------------------------------------------------------
// Kernel 5: inverse DFT as WMMA GEMM + overlap-add + event sum via f32 atomics
// ---------------------------------------------------------------------------
__global__ __launch_bounds__(256)
void k_irfft_ola(const _Float16* __restrict__ Afin, const _Float16* __restrict__ BirfF,
                 float* __restrict__ sig) {
    __shared__ _Float16 ldsA[16 * LDSI_STR];
    const int r0  = blockIdx.x * 16;
    const int tid = threadIdx.x;
    // vectorized 16B staging: global_load_b128 -> ds_store_b128
    for (int idx8 = tid; idx8 < 16 * (NB2 / 8); idx8 += 256) {
        int m  = idx8 / (NB2 / 8);
        int j8 = (idx8 % (NB2 / 8)) * 8;
        *(v8h*)(ldsA + m * LDSI_STR + j8) =
            *(const v8h*)(Afin + ((size_t)(r0 + m) * NB2 + j8));
    }
    __syncthreads();

    const int lane = tid & 31, wave = tid >> 5;
    const int Mrow = lane & 15;
    const int hiA  = (lane >= 16) ? 8 : 0;
    const int hiM  = (lane >= 16) ? 8 : 0;
    const _Float16* ldsRow = ldsA + Mrow * LDSI_STR + hiA;

    v16h aF[KB_INV];
#pragma unroll
    for (int kb = 0; kb < KB_INV; ++kb) aF[kb] = load_a_frag(ldsRow + kb * 32);

    for (int nt = wave; nt < NT_INV; nt += 8) {
        const _Float16* bptr = BirfF + ((size_t)nt * KB_INV) * FRAG + lane * 16;
        __builtin_prefetch(bptr + (size_t)8 * KB_INV * FRAG, 0, 3);
        v8f c = wmma_tile<KB_INV>(aF, bptr);
        const int j = nt * 16 + Mrow;
#pragma unroll
        for (int v = 0; v < 8; ++v) {
            int r  = r0 + v + hiM;
            int f  = r & (N_FRAMES - 1);
            int bb = r / (N_FRAMES * N_EVENTS);
            atomicAdd(&sig[(size_t)bb * SIGPAD + f * STEP + j], c[v]);
        }
    }
}

// ---------------------------------------------------------------------------
// Kernel 6: per-batch max-norm
// ---------------------------------------------------------------------------
__global__ __launch_bounds__(512)
void k_norm(const float* __restrict__ sig, float* __restrict__ out) {
    __shared__ float red[17];
    const int b = blockIdx.x, tid = threadIdx.x;
    const float* s = sig + (size_t)b * SIGPAD;
    float mx = 0.f;
    for (int t = tid; t < N_SAMPLES; t += blockDim.x) mx = fmaxf(mx, fabsf(s[t]));
    for (int o = 16; o > 0; o >>= 1) mx = fmaxf(mx, __shfl_xor(mx, o, 32));
    if ((tid & 31) == 0) red[tid >> 5] = mx;
    __syncthreads();
    if (tid < 16) {
        float v = red[tid];
        for (int o = 8; o > 0; o >>= 1) v = fmaxf(v, __shfl_xor(v, o, 32));
        if (tid == 0) red[16] = v;
    }
    __syncthreads();
    float inv = 1.f / (red[16] + 1e-8f);
    for (int t = tid; t < N_SAMPLES; t += blockDim.x)
        out[(size_t)b * N_SAMPLES + t] = s[t] * inv;
}

// ---------------------------------------------------------------------------
extern "C" void kernel_launch(void* const* d_in, const int* in_sizes, int n_in,
                              void* d_out, int out_size, void* d_ws, size_t ws_size,
                              hipStream_t stream) {
    (void)in_sizes; (void)n_in; (void)out_size; (void)ws_size;
    const float* x     = (const float*)d_in[0];   // (32,16,533) f32
    const float* noise = (const float*)d_in[1];   // (32,16,32768) f32
    float* out = (float*)d_out;                   // (32,1,32768) f32

    char* ws = (char*)d_ws;
    size_t off = 0;
    float*    atoms = (float*)   (ws + off); off += (size_t)NROWS * N_SAMPLES * 4;        // 64 MB
    _Float16* BfwdF = (_Float16*)(ws + off); off += (size_t)NT_FWD * KB_FWD * FRAG * 2;   // 544 KB
    _Float16* BirfF = (_Float16*)(ws + off); off += (size_t)NT_INV * KB_INV * FRAG * 2;   // 544 KB
    float*    specP = (float*)   (ws + off); off += (size_t)NFR_TOT * NB2 * 4;            // 143 MB
    _Float16* Afin  = (_Float16*)(ws + off); off += (size_t)NFR_TOT * NB2 * 2;            // 68 MB
    float*    sig   = (float*)   (ws + off); off += (size_t)NBATCH * SIGPAD * 4;          // 4 MB

    const int tabTotal = NT_FWD * KB_FWD * FRAG + NT_INV * KB_INV * FRAG;
    k_build_tables<<<(tabTotal + 255) / 256, 256, 0, stream>>>(BfwdF, BirfF);
    k_zero<<<(NBATCH * SIGPAD + 255) / 256, 256, 0, stream>>>(sig, NBATCH * SIGPAD);
    k_noise_atoms<<<NROWS, 512, 2 * N2 * sizeof(float), stream>>>(x, noise, atoms);
    k_stft<<<NFR_TOT / 16, 256, 0, stream>>>(atoms, BfwdF, specP);
    k_scan<<<(NROWS * NPAD + 255) / 256, 256, 0, stream>>>(x, specP, Afin);
    k_irfft_ola<<<NFR_TOT / 16, 256, 0, stream>>>(Afin, BirfF, sig);
    k_norm<<<NBATCH, 512, 0, stream>>>(sig, out);
}
